// LLM_22351009809300
// MI455X (gfx1250) — matrix operand order repaired and checked by reading, not compiled, not run on hardware
//
#include <hip/hip_runtime.h>
#include <hip/hip_bf16.h>
#include <stdint.h>

// ---------------------------------------------------------------------------
// LLM sampling decode step on MI455X (gfx1250).
// Bandwidth-bound: ~105MB traffic -> ~4.5us floor @ 23.3 TB/s.
// 51.5MB logits fit in 192MB L2 -> multi-pass exact radix-select top-k is
// cheap (1 HBM pass + 3 L2 passes). CDNA5 paths used:
//   * tensor_load_to_lds (TDM) + s_wait_tensorcnt : DMA generated-token row
//   * LDS atomics (ds_*) for presence bitmap + histograms
//   * global_prefetch_b8 via __builtin_prefetch on the streaming pass
//   * v_wmma_f32_16x16x32_f16 for the top-p cumulative-sum over sorted
//     survivors (triangular 0/1 A-matrix, f32 accumulation)
// This revision: b128 vectorized streaming/zero-fill (issue-rate bound at
// 23.3TB/s otherwise) and O(log N) parallel bin-select instead of serial walks.
// ---------------------------------------------------------------------------

#define TEMPERATURE   0.8f
#define TOP_K         50
#define TOP_P         0.9f
#define REP_PENALTY   1.2f

#define BATCH     256
#define VCAP      50272            // padded vocab capacity (multiple of 32)
#define TCAP      2048
#define NTHREADS  256
#define SURV_CAP  128              // survivor buffer (top-k=50 + tie slack)

typedef __attribute__((ext_vector_type(16))) _Float16 v16h;
typedef __attribute__((ext_vector_type(8)))  float    v8f;
typedef unsigned int __attribute__((ext_vector_type(4))) u32x4;
typedef int          __attribute__((ext_vector_type(4))) i32x4;
typedef int          __attribute__((ext_vector_type(8))) i32x8;

// Order-preserving float -> uint key (larger float => larger key).
__device__ __forceinline__ unsigned fkey(float x) {
  unsigned u = __float_as_uint(x);
  return (u & 0x80000000u) ? ~u : (u | 0x80000000u);
}

__device__ __forceinline__ unsigned hash3(unsigned a, unsigned b, unsigned c) {
  unsigned h = a * 0x9E3779B9u ^ b * 0x85EBCA6Bu ^ c * 0xC2B2AE35u;
  h ^= h >> 16; h *= 0x7FEB352Du; h ^= h >> 15; h *= 0x846CA68Bu; h ^= h >> 16;
  return h;
}

// Stream a 4B-aligned float row with a b128 body (head/tail scalar to reach
// 16B alignment; V is odd so rows are only dword-aligned).
template <typename F>
__device__ __forceinline__ void stream_row(const float* __restrict__ row, int V,
                                           int tid, F&& f) {
  int lead = (int)((4u - (((uintptr_t)row >> 2) & 3u)) & 3u);
  if (lead > V) lead = V;
  if (tid < lead) f(tid, row[tid]);
  const int nv = (V - lead) >> 2;
  const float4* __restrict__ row4 = (const float4*)(row + lead);
  for (int i = tid; i < nv; i += NTHREADS) {
    __builtin_prefetch(row4 + i + 4 * NTHREADS, 0, 0);  // global_prefetch_b8
    float4 q = row4[i];
    const int j = lead + 4 * i;
    f(j + 0, q.x); f(j + 1, q.y); f(j + 2, q.z); f(j + 3, q.w);
  }
  for (int i = lead + 4 * nv + tid; i < V; i += NTHREADS) f(i, row[i]);
}

// Parallel descending radix-bin select over `nbins` LDS histogram bins.
// On entry *s_need holds k (count from the top). On exit *s_bin is the bin
// containing the k-th largest key and *s_need the residual count within it.
// suffix(bin) = sum_{j>=bin} hist[j]; pick unique bin with
// suffix(bin) >= need > suffix(bin+1). All NTHREADS threads participate.
__device__ __forceinline__ void select_bin_desc(const unsigned* hist, int nbins,
                                                unsigned* red_u,
                                                unsigned* s_bin, int* s_need,
                                                int tid) {
  const int per  = nbins / NTHREADS;          // 8 (2048 bins) or 4 (1024 bins)
  const int base = tid * per;
  unsigned loc[8];
  unsigned tot = 0u;
  for (int r = 0; r < per; ++r) { loc[r] = hist[base + r]; tot += loc[r]; }
  red_u[tid] = tot;
  __syncthreads();
  // Hillis-Steele inclusive *suffix* scan of per-thread totals.
  for (int off = 1; off < NTHREADS; off <<= 1) {
    unsigned add = (tid + off < NTHREADS) ? red_u[tid + off] : 0u;
    __syncthreads();
    red_u[tid] += add;
    __syncthreads();
  }
  const unsigned need = (unsigned)*s_need;
  __syncthreads();                            // all reads of need done before winner writes
  unsigned suf_next = red_u[tid] - tot;       // suffix(base + per) = sum over threads > tid
  for (int r = per - 1; r >= 0; --r) {
    unsigned suf = suf_next + loc[r];
    if (suf >= need && suf_next < need) {     // unique boundary bin
      *s_bin  = (unsigned)(base + r);
      *s_need = (int)(need - suf_next);
    }
    suf_next = suf;
  }
  __syncthreads();
}

__global__ __launch_bounds__(NTHREADS)
void llm_sample_kernel(const float* __restrict__ logits,
                       const int*  __restrict__ generated,
                       float* __restrict__ out, int V, int T)
{
  const int b   = blockIdx.x;
  const int tid = threadIdx.x;

  __shared__ unsigned bitmap[VCAP / 32];   // repetition-penalty presence bits
  __shared__ int      genbuf[TCAP];        // TDM destination
  __shared__ unsigned hist[2048];          // radix-select histogram (reused)
  __shared__ float    red[NTHREADS];
  __shared__ unsigned red_u[NTHREADS];
  __shared__ float    sv_val[SURV_CAP];
  __shared__ int      sv_idx[SURV_CAP];
  __shared__ float    es[SURV_CAP];        // exp(x - max) numerators
  __shared__ float    parr[SURV_CAP];      // softmax probs over survivors
  __shared__ float    cum[SURV_CAP];       // inclusive prefix of parr (sorted)
  __shared__ int      keepf[SURV_CAP];     // top-p keep flags
  __shared__ float    cmat[16 * 16];       // WMMA D-fragment spill
  __shared__ unsigned s_cnt, s_bin;
  __shared__ int      s_need;
  __shared__ float    s_max, s_esum, s_zk;
  __shared__ int      s_tok;

  // ---- init ---------------------------------------------------------------
  for (int i = tid; i < VCAP / 32; i += NTHREADS) bitmap[i] = 0u;
  if (tid == 0) { s_cnt = 0u; s_need = TOP_K; }
  __syncthreads();

  // ---- stage generated[b,:] into LDS via the Tensor Data Mover ------------
#if __has_builtin(__builtin_amdgcn_tensor_load_to_lds) && __has_builtin(__builtin_amdgcn_s_wait_tensorcnt)
  if (tid < 32) {  // one wave issues one TDM op; TENSORcnt is per-wave
    // D# group 0: count=1 | lds_addr | global_addr[56:0] | type=2
    unsigned lds_off = (unsigned)(uintptr_t)&genbuf[0];  // addr[31:0] == LDS byte offset
    unsigned long long ga = (unsigned long long)(uintptr_t)(generated + (size_t)b * T);
    u32x4 g0;
    g0.x = 1u;                                              // count=1, user mode
    g0.y = lds_off;                                         // lds_addr (bytes)
    g0.z = (unsigned)(ga & 0xFFFFFFFFu);                    // global_addr[31:0]
    g0.w = (unsigned)((ga >> 32) & 0x01FFFFFFu) | (2u << 30); // addr[56:32] | type=2
    // D# group 1: data_size=4B, 1-D tensor/tile of T int32 elements
    i32x8 g1;
    g1[0] = (int)(2u << 16);                                // data_size code 2 (4B), mask=0
    g1[1] = (int)(((unsigned)T & 0xFFFFu) << 16);           // tensor_dim0[15:0] @ bits 63:48
    g1[2] = (int)((((unsigned)T >> 16) & 0xFFFFu) | (1u << 16)); // dim0 hi | tensor_dim1=1
    g1[3] = (int)(((unsigned)T & 0xFFFFu) << 16);           // tile_dim0=T @ bits 127:112
    g1[4] = 0;                                              // tile_dim1=0, tile_dim2=0
    g1[5] = (int)(unsigned)T;                               // tensor_dim0_stride lo32
    g1[6] = 0;
    g1[7] = 0;
    i32x4 gz = {0, 0, 0, 0};
  #if defined(__clang_major__) && (__clang_major__ >= 23)
    i32x8 gz8 = {0, 0, 0, 0, 0, 0, 0, 0};
    __builtin_amdgcn_tensor_load_to_lds(g0, g1, gz, gz, gz8, 0);
  #else
    __builtin_amdgcn_tensor_load_to_lds(g0, g1, gz, gz, 0);
  #endif
    __builtin_amdgcn_s_wait_tensorcnt(0);
  }
#else
  for (int t = tid; t < T; t += NTHREADS) genbuf[t] = generated[(size_t)b * T + t];
#endif
  __syncthreads();

  // ---- presence bitmap (LDS atomics) --------------------------------------
  for (int t = tid; t < T; t += NTHREADS) {
    int v = genbuf[t];
    if (v >= 0 && v < V) atomicOr(&bitmap[v >> 5], 1u << (v & 31));
  }
  __syncthreads();

  const float invT  = 1.0f / TEMPERATURE;
  const float invTR = 1.0f / (TEMPERATURE * REP_PENALTY);
  const float* row  = logits + (size_t)b * V;
  auto scale = [&](int i, float raw) -> float {  // temperature + rep. penalty
    bool pen = (bitmap[i >> 5] >> (i & 31)) & 1u;
    return raw * (pen ? invTR : invT);
  };

  // ---- pass 1 (HBM, b128): row max + histogram of key bits [31:21] --------
  for (int i = tid; i < 2048; i += NTHREADS) hist[i] = 0u;
  __syncthreads();
  float lmax = -INFINITY;
  stream_row(row, V, tid, [&](int i, float raw) {
    float x = scale(i, raw);
    lmax = fmaxf(lmax, x);
    atomicAdd(&hist[fkey(x) >> 21], 1u);
  });
  red[tid] = lmax;
  __syncthreads();
  for (int s = NTHREADS / 2; s > 0; s >>= 1) {
    if (tid < s) red[tid] = fmaxf(red[tid], red[tid + s]);
    __syncthreads();
  }
  if (tid == 0) s_max = red[0];
  select_bin_desc(hist, 2048, red_u, &s_bin, &s_need, tid);
  const unsigned b0 = s_bin;   // safe: next writer syncs before touching s_bin

  // ---- pass 2 (L2, b128): refine bits [20:10] -----------------------------
  for (int i = tid; i < 2048; i += NTHREADS) hist[i] = 0u;
  __syncthreads();
  stream_row(row, V, tid, [&](int i, float raw) {
    unsigned k = fkey(scale(i, raw));
    if ((k >> 21) == b0) atomicAdd(&hist[(k >> 10) & 2047u], 1u);
  });
  __syncthreads();
  select_bin_desc(hist, 2048, red_u, &s_bin, &s_need, tid);
  const unsigned b1 = s_bin;

  // ---- pass 3 (L2, b128): refine bits [9:0] -> exact 50th-largest key -----
  for (int i = tid; i < 2048; i += NTHREADS) hist[i] = 0u;
  __syncthreads();
  const unsigned pref21 = (b0 << 11) | b1;
  stream_row(row, V, tid, [&](int i, float raw) {
    unsigned k = fkey(scale(i, raw));
    if ((k >> 10) == pref21) atomicAdd(&hist[k & 1023u], 1u);
  });
  __syncthreads();
  select_bin_desc(hist, 1024, red_u, &s_bin, &s_need, tid);
  const unsigned kthkey = (b0 << 21) | (b1 << 10) | s_bin;

  if (tid < SURV_CAP) { sv_val[tid] = -INFINITY; sv_idx[tid] = 0; }
  __syncthreads();

  // ---- pass 4 (L2, b128): collect survivors (x >= kth, ties kept) ---------
  stream_row(row, V, tid, [&](int i, float raw) {
    float x = scale(i, raw);
    if (fkey(x) >= kthkey) {
      unsigned slot = atomicAdd(&s_cnt, 1u);
      if (slot < SURV_CAP) { sv_val[slot] = x; sv_idx[slot] = i; }
    }
  });
  __syncthreads();
  const int n = min((int)s_cnt, SURV_CAP);

  // ---- bitonic sort survivors descending by value (128 padded slots) ------
  for (int k = 2; k <= SURV_CAP; k <<= 1) {
    for (int j = k >> 1; j > 0; j >>= 1) {
      __syncthreads();
      if (tid < SURV_CAP) {
        int ixj = tid ^ j;
        if (ixj > tid) {
          bool up = ((tid & k) == 0);
          float a = sv_val[tid], bb = sv_val[ixj];
          bool sw = up ? (a < bb) : (a > bb);   // descending overall
          if (sw) {
            sv_val[tid] = bb; sv_val[ixj] = a;
            int ti = sv_idx[tid]; sv_idx[tid] = sv_idx[ixj]; sv_idx[ixj] = ti;
          }
        }
      }
    }
  }
  __syncthreads();

  // ---- softmax numerators & probs over survivors --------------------------
  if (tid < SURV_CAP) es[tid] = (tid < n) ? __expf(sv_val[tid] - s_max) : 0.0f;
  __syncthreads();
  if (tid == 0) { float s = 0.f; for (int i = 0; i < n; ++i) s += es[i]; s_esum = s; }
  __syncthreads();
  if (tid < SURV_CAP) parr[tid] = es[tid] / s_esum;
  __syncthreads();

  // ---- top-p cumulative sum via V_WMMA_F32_16X16X32_F16 -------------------
  // A(16x32) = lower-triangular 0/1 masks: A[m][k] = (k <= 2m+1)  (exact in f16)
  // B(32x16): col 0 = parr[0..31], col 1 = parr[32..63], other cols 0.
  // C[m][col] = inclusive prefix at odd position 2m+1 of that half (f32 acc).
  if (tid < 32) {                       // full wave 0, EXEC all ones
    const int lane = tid;
    v16h a, bm;
    for (int j = 0; j < 16; ++j) {
      // A layout (16-bit A 16x32): lanes 0-15 rows M=lane, K in {0..7,16..23};
      // lanes 16-31 same rows, K in {8..15,24..31}.
      int K = ((j < 8) ? j : (j + 8)) + ((lane < 16) ? 0 : 8);
      int M = lane & 15;
      a[j] = (_Float16)((K <= 2 * M + 1) ? 1.0f : 0.0f);
      // B layout (32x16): same K order as A: lanes 0-15 K in {0..7,16..23}; lanes 16-31 K in {8..15,24..31}, N=lane%16.
      int Kb = ((j < 8) ? j : (j + 8)) + ((lane < 16) ? 0 : 8);
      int N  = lane & 15;
      float bv = (N == 0) ? parr[Kb] : ((N == 1) ? parr[32 + Kb] : 0.0f);
      bm[j] = (_Float16)bv;
    }
    v8f c = {};
    c = __builtin_amdgcn_wmma_f32_16x16x32_f16(false, a, false, bm,
                                               (short)0, c, false, false);
    // D layout: lane L, reg r -> C[r + 8*(L/16)][L%16]
    for (int r = 0; r < 8; ++r)
      cmat[(r + 8 * (lane >> 4)) * 16 + (lane & 15)] = c[r];
  }
  __syncthreads();

  if (tid == 0) {
    float half0 = cmat[15 * 16 + 0];           // sum of parr[0..31]
    cum[0] = parr[0];
    for (int m = 0; m < 16; ++m) {
      cum[2 * m + 1] = cmat[m * 16 + 0];       // odd positions, half 0
      int i0 = 2 * m + 2;                      // even fixup: pre(2m+2)=C[m][0]+p
      if (i0 < 32) cum[i0] = cmat[m * 16 + 0] + parr[i0];
    }
    cum[32] = half0 + parr[32];
    for (int m = 0; m < 16; ++m) {
      cum[32 + 2 * m + 1] = half0 + cmat[m * 16 + 1];
      int i0 = 32 + 2 * m + 2;
      if (i0 < 64) cum[i0] = half0 + cmat[m * 16 + 1] + parr[i0];
    }
    for (int i = 64; i < n; ++i) cum[i] = cum[i - 1] + parr[i];  // tie overflow tail

    // keep while cumulative mass <= p; always keep rank 0
    float zk = 0.f;
    for (int i = 0; i < SURV_CAP; ++i) {
      int kp = (i < n) && ((i == 0) || (cum[i] <= TOP_P));
      keepf[i] = kp;
      if (kp) zk += es[i];
    }
    s_zk = zk;

    // Gumbel-max categorical sample over the kept set (deterministic PRNG)
    float best = -INFINITY; int btok = sv_idx[0];
    for (int i = 0; i < n; ++i) {
      if (!keepf[i]) continue;
      unsigned h = hash3(42u, (unsigned)b, (unsigned)sv_idx[i]);
      float u = ((float)h + 0.5f) * (1.0f / 4294967296.0f);
      float g = -__logf(-__logf(u));
      float sc = sv_val[i] + g;
      if (sc > best) { best = sc; btok = sv_idx[i]; }
    }
    s_tok = btok;
  }
  __syncthreads();

  // ---- outputs: next_token[B] then probs[B,V] (b128 zero-fill + scatter) --
  float* tok_out = out;
  float* prow    = out + BATCH + (size_t)b * V;
  if (tid == 0) tok_out[b] = (float)s_tok;
  {
    int lead = (int)((4u - (((uintptr_t)prow >> 2) & 3u)) & 3u);
    if (lead > V) lead = V;
    if (tid < lead) prow[tid] = 0.0f;
    const int nv = (V - lead) >> 2;
    float4* p4 = (float4*)(prow + lead);
    const float4 z4 = make_float4(0.f, 0.f, 0.f, 0.f);
    for (int i = tid; i < nv; i += NTHREADS) p4[i] = z4;
    for (int i = lead + 4 * nv + tid; i < V; i += NTHREADS) prow[i] = 0.0f;
  }
  __syncthreads();  // order zero-fill before sparse overwrite (block scope)
  if (tid < SURV_CAP && keepf[tid]) prow[sv_idx[tid]] = es[tid] / s_zk;
}

extern "C" void kernel_launch(void* const* d_in, const int* in_sizes, int n_in,
                              void* d_out, int out_size, void* d_ws, size_t ws_size,
                              hipStream_t stream) {
  const float* logits = (const float*)d_in[0];
  const int*   gen    = (const int*)d_in[1];
  float* out = (float*)d_out;
  int V = in_sizes[0] / BATCH;   // 50257
  int T = in_sizes[1] / BATCH;   // 2048
  llm_sample_kernel<<<BATCH, NTHREADS, 0, stream>>>(logits, gen, out, V, T);
}
